// PK_61821759259201
// MI455X (gfx1250) — compile-verified, matrix operand order checked by reading, not run on hardware
//
#include <hip/hip_runtime.h>
#include <hip/hip_bf16.h>

// ---------------- problem constants (from reference) ----------------
#define B_   4
#define S_   2048
#define D_   1024
#define P_   2
#define K_   1000
#define KP_  1008          // K padded to multiple of 16
#define H_   8
#define DK_  512
#define BS_  8192          // B*S
#define QW_  8192          // P*H*DK (q width)
#define NT_  63            // KP_/16 n-tiles
#define TK_  16            // top-k

typedef __bf16 bf16;
typedef __attribute__((ext_vector_type(16))) __bf16 v16bf;
typedef __attribute__((ext_vector_type(8)))  float  v8f;

__device__ __forceinline__ bf16 f2bf(float f) {
    unsigned int u = __builtin_bit_cast(unsigned int, f);
    unsigned int r = (u + 0x7FFFu + ((u >> 16) & 1u)) >> 16;   // round-to-nearest-even
    return __builtin_bit_cast(bf16, (unsigned short)r);
}

#define WMMA_BF16(A, B, C) \
    __builtin_amdgcn_wmma_f32_16x16x32_bf16(false, (A), false, (B), (short)0, (C), false, false)

// ---------------- prep kernels ----------------
__global__ void __launch_bounds__(256) k_cvt_x(const float* __restrict__ x, bf16* __restrict__ xb) {
    size_t i = (size_t)blockIdx.x * 256 + threadIdx.x;   // BS_*D_ total
    xb[i] = f2bf(x[i]);
}

// w_q is [D_ x QW_]; produce wqT bf16 [QW_ x D_] so contraction is contiguous
__global__ void __launch_bounds__(256) k_cvt_wqt(const float* __restrict__ wq, bf16* __restrict__ wqt) {
    size_t i = (size_t)blockIdx.x * 256 + threadIdx.x;   // QW_*D_ total
    int n = (int)(i / D_);
    int d = (int)(i % D_);
    wqt[i] = f2bf(wq[(size_t)d * QW_ + n]);
}

// keys [P,K,H,DK] f32 -> keysb [P,H,KP_,DK] bf16, pad rows (k>=K_) zeroed
__global__ void __launch_bounds__(256) k_cvt_keys(const float* __restrict__ keys, bf16* __restrict__ keysb) {
    size_t i = (size_t)blockIdx.x * 256 + threadIdx.x;   // P_*H_*KP_*DK_ total
    int d  = (int)(i % DK_);
    size_t r = i / DK_;
    int kk = (int)(r % KP_);
    size_t r2 = r / KP_;
    int h = (int)(r2 % H_);
    int p = (int)(r2 / H_);
    float v = 0.0f;
    if (kk < K_) v = keys[(((size_t)p * K_ + kk) * H_ + h) * DK_ + d];
    keysb[i] = f2bf(v);
}

// ---------------- GEMM1: q[BS_ x QW_] = xb[BS_ x D_] @ wqT^T ----------------
// one wave computes a 32(M) x 64(N) tile; 256 mblk x 128 nstrip = 32768 waves.
// Software-pipelined: iteration k's 8 WMMAs consume fragments loaded at k-1,
// while the 6 loads for k+1 are already in flight (keeps loadcnt slack > 0).
__global__ void __launch_bounds__(256) k_gemm_q(const bf16* __restrict__ xb,
                                                const bf16* __restrict__ wqt,
                                                bf16* __restrict__ qb) {
    int lane = threadIdx.x & 31;
    int wv   = blockIdx.x * 8 + (threadIdx.x >> 5);
    int mblk   = wv >> 7;          // 0..255
    int nstrip = wv & 127;         // 0..127
    int m0 = mblk * 32;
    int n0 = nstrip * 64;
    int rsel = lane & 15;
    int ksel = (lane >> 4) * 16;

    const bf16* a0r = xb + (size_t)(m0 +  0 + rsel) * D_ + ksel;
    const bf16* a1r = xb + (size_t)(m0 + 16 + rsel) * D_ + ksel;
    const bf16* b0r = wqt + (size_t)(n0 +  0 + rsel) * D_ + ksel;
    const bf16* b1r = wqt + (size_t)(n0 + 16 + rsel) * D_ + ksel;
    const bf16* b2r = wqt + (size_t)(n0 + 32 + rsel) * D_ + ksel;
    const bf16* b3r = wqt + (size_t)(n0 + 48 + rsel) * D_ + ksel;

    v8f acc00 = {}, acc01 = {}, acc02 = {}, acc03 = {};
    v8f acc10 = {}, acc11 = {}, acc12 = {}, acc13 = {};

    // prologue: fragments for k = 0
    v16bf a0 = *(const v16bf*)(a0r);
    v16bf a1 = *(const v16bf*)(a1r);
    v16bf b0 = *(const v16bf*)(b0r);
    v16bf b1 = *(const v16bf*)(b1r);
    v16bf b2 = *(const v16bf*)(b2r);
    v16bf b3 = *(const v16bf*)(b3r);

#pragma unroll 1
    for (int k = 32; k < D_; k += 32) {
        // prefetch next k-step into fresh registers
        v16bf na0 = *(const v16bf*)(a0r + k);
        v16bf na1 = *(const v16bf*)(a1r + k);
        v16bf nb0 = *(const v16bf*)(b0r + k);
        v16bf nb1 = *(const v16bf*)(b1r + k);
        v16bf nb2 = *(const v16bf*)(b2r + k);
        v16bf nb3 = *(const v16bf*)(b3r + k);
        // consume current fragments
        acc00 = WMMA_BF16(a0, b0, acc00);
        acc10 = WMMA_BF16(a1, b0, acc10);
        acc01 = WMMA_BF16(a0, b1, acc01);
        acc11 = WMMA_BF16(a1, b1, acc11);
        acc02 = WMMA_BF16(a0, b2, acc02);
        acc12 = WMMA_BF16(a1, b2, acc12);
        acc03 = WMMA_BF16(a0, b3, acc03);
        acc13 = WMMA_BF16(a1, b3, acc13);
        a0 = na0; a1 = na1; b0 = nb0; b1 = nb1; b2 = nb2; b3 = nb3;
    }
    // epilogue: last k-step
    acc00 = WMMA_BF16(a0, b0, acc00);
    acc10 = WMMA_BF16(a1, b0, acc10);
    acc01 = WMMA_BF16(a0, b1, acc01);
    acc11 = WMMA_BF16(a1, b1, acc11);
    acc02 = WMMA_BF16(a0, b2, acc02);
    acc12 = WMMA_BF16(a1, b2, acc12);
    acc03 = WMMA_BF16(a0, b3, acc03);
    acc13 = WMMA_BF16(a1, b3, acc13);

    // D layout: vgpr i, lanes 0-15: M=i, N=lane; lanes 16-31: M=8+i, N=lane-16
    int rb0 = m0 + (lane >> 4) * 8;
    int col = lane & 15;
#pragma unroll
    for (int i = 0; i < 8; ++i) {
        size_t ro0 = (size_t)(rb0 + i) * QW_ + n0 + col;        // rows m0..m0+15
        size_t ro1 = ro0 + (size_t)16 * QW_;                    // rows m0+16..m0+31
        qb[ro0 +  0] = f2bf(acc00[i]);
        qb[ro0 + 16] = f2bf(acc01[i]);
        qb[ro0 + 32] = f2bf(acc02[i]);
        qb[ro0 + 48] = f2bf(acc03[i]);
        qb[ro1 +  0] = f2bf(acc10[i]);
        qb[ro1 + 16] = f2bf(acc11[i]);
        qb[ro1 + 32] = f2bf(acc12[i]);
        qb[ro1 + 48] = f2bf(acc13[i]);
    }
}

// ---------------- GEMM2 + fused top-16 ----------------
// one wave per (32-row block, p*8+h): 256 * 16 = 4096 waves.
// Per n-tile: 2 stacked 16x16 D-tiles share each keys B-fragment; every lane
// then owns one of the 32 rows for the register-resident sorted top-16 scan.
__global__ void __launch_bounds__(256) k_sim_topk(const bf16* __restrict__ qb,
                                                  const bf16* __restrict__ keysb,
                                                  float* __restrict__ tks,
                                                  int*   __restrict__ tki) {
    __shared__ float tile[8][512];        // per-wave slab: two 16x16 f32 tiles
    int lane  = threadIdx.x & 31;
    int wslot = threadIdx.x >> 5;
    int wv = blockIdx.x * 8 + wslot;      // 0..4095
    int ph = wv & 15;                     // p*8+h
    int rb = wv >> 4;                     // 0..255
    int p  = ph >> 3;
    int h  = ph & 7;
    int r0 = rb * 32;
    int rsel = lane & 15;
    int ksel = (lane >> 4) * 16;

    const bf16* a0base = qb + (size_t)(r0 +  0 + rsel) * QW_ + ph * DK_ + ksel;
    const bf16* a1base = qb + (size_t)(r0 + 16 + rsel) * QW_ + ph * DK_ + ksel;
    const bf16* bbase  = keysb + (size_t)ph * KP_ * DK_ + ksel;

    float ts[TK_]; int ti[TK_];
#pragma unroll
    for (int j = 0; j < TK_; ++j) { ts[j] = -3.0e38f; ti[j] = 0; }

#pragma unroll 1
    for (int nt = 0; nt < NT_; ++nt) {
        int n0 = nt * 16;
        const bf16* bb = bbase + (size_t)(n0 + rsel) * DK_;
        v8f acc0 = {}, acc1 = {};
        // software-pipelined contraction over DK_ (16 steps of 32)
        v16bf a0 = *(const v16bf*)(a0base);
        v16bf a1 = *(const v16bf*)(a1base);
        v16bf b  = *(const v16bf*)(bb);
#pragma unroll 1
        for (int k = 32; k < DK_; k += 32) {
            v16bf na0 = *(const v16bf*)(a0base + k);
            v16bf na1 = *(const v16bf*)(a1base + k);
            v16bf nb  = *(const v16bf*)(bb + k);
            acc0 = WMMA_BF16(a0, b, acc0);
            acc1 = WMMA_BF16(a1, b, acc1);
            a0 = na0; a1 = na1; b = nb;
        }
        acc0 = WMMA_BF16(a0, b, acc0);
        acc1 = WMMA_BF16(a1, b, acc1);

        // stage both D-tiles: tile0 rows 0-15 at [0..255], tile1 rows 16-31 at [256..511]
        int rl = (lane >> 4) * 8;
#pragma unroll
        for (int i = 0; i < 8; ++i) {
            tile[wslot][(rl + i) * 16 + (lane & 15)]       = acc0[i];
            tile[wslot][256 + (rl + i) * 16 + (lane & 15)] = acc1[i];
        }
        // wave32 lockstep + in-order DS: no barrier needed (per-wave LDS slab).
        // lane L owns local row L; its row lives at offset L*16 in the slab
        // (256 + (L-16)*16 == L*16 for L>=16).
#pragma unroll 1
        for (int c = 0; c < 16; ++c) {
            float v = tile[wslot][lane * 16 + c];
            int idx = n0 + c;
            if (idx < K_ && v > ts[TK_ - 1]) {
#pragma unroll
                for (int j = 0; j < TK_; ++j) {
                    if (v > ts[j]) {
                        float tf = ts[j]; ts[j] = v; v = tf;
                        int   tt = ti[j]; ti[j] = idx; idx = tt;
                    }
                }
            }
        }
    }
    size_t ob = ((size_t)(p * BS_ + r0 + lane) * H_ + h) * TK_;
#pragma unroll
    for (int j = 0; j < TK_; ++j) { tks[ob + j] = ts[j]; tki[ob + j] = ti[j]; }
}

// ---------------- merge: sorted16 x sorted16 -> top16 ----------------
__global__ void __launch_bounds__(256) k_merge(const float* __restrict__ tks,
                                               const int*   __restrict__ tki,
                                               float* __restrict__ outs,
                                               float* __restrict__ outi) {
    int t = blockIdx.x * 256 + threadIdx.x;    // 0..BS_*H_-1  ( = bs*H + h )
    const float* sa = tks + (size_t)t * TK_;
    const int*   ia = tki + (size_t)t * TK_;
    const float* sb = tks + ((size_t)BS_ * H_ + t) * TK_;
    const int*   ib = tki + ((size_t)BS_ * H_ + t) * TK_;

    float ts[TK_]; int ti[TK_];
#pragma unroll
    for (int j = 0; j < TK_; ++j) { ts[j] = -3.0e38f; ti[j] = 0; }

    float sb0 = sb[0];
#pragma unroll 1
    for (int i = 0; i < TK_; ++i) {
        float base = sa[i];
        if (base + sb0 <= ts[TK_ - 1]) break;
        int iai = ia[i];
#pragma unroll 1
        for (int j = 0; j < TK_; ++j) {
            float v = base + sb[j];
            if (v <= ts[TK_ - 1]) break;
            int idx = iai + K_ * ib[j];
#pragma unroll
            for (int q = 0; q < TK_; ++q) {
                if (v > ts[q]) {
                    float tf = ts[q]; ts[q] = v; v = tf;
                    int   tt = ti[q]; ti[q] = idx; idx = tt;
                }
            }
        }
    }
    size_t ob = (size_t)t * TK_;
#pragma unroll
    for (int j = 0; j < TK_; ++j) { outs[ob + j] = ts[j]; outi[ob + j] = (float)ti[j]; }
}

// ---------------- launcher ----------------
extern "C" void kernel_launch(void* const* d_in, const int* in_sizes, int n_in,
                              void* d_out, int out_size, void* d_ws, size_t ws_size,
                              hipStream_t stream) {
    const float* x    = (const float*)d_in[0];
    const float* wq   = (const float*)d_in[1];
    const float* keys = (const float*)d_in[2];

    // workspace layout (bytes, 256-aligned); total ~192 MB
    char* ws = (char*)d_ws;
    size_t o = 0;
    bf16* xb    = (bf16*)(ws + o); o += (size_t)BS_ * D_ * 2;            // 16 MB
    bf16* wqt   = (bf16*)(ws + o); o += (size_t)QW_ * D_ * 2;            // 16 MB
    bf16* keysb = (bf16*)(ws + o); o += (size_t)P_ * H_ * KP_ * DK_ * 2; // ~15.75 MB
    bf16* qb    = (bf16*)(ws + o); o += (size_t)BS_ * QW_ * 2;           // 128 MB
    float* tks  = (float*)(ws + o); o += (size_t)P_ * BS_ * H_ * TK_ * 4; // 8 MB
    int*   tki  = (int*)(ws + o);

    float* outs = (float*)d_out;
    float* outi = outs + (size_t)BS_ * H_ * TK_;

    k_cvt_x   <<<(BS_ * D_) / 256, 256, 0, stream>>>(x, xb);
    k_cvt_wqt <<<(QW_ * D_) / 256, 256, 0, stream>>>(wq, wqt);
    k_cvt_keys<<<(P_ * H_ * KP_ * DK_) / 256, 256, 0, stream>>>(keys, keysb);
    k_gemm_q  <<<4096, 256, 0, stream>>>(xb, wqt, qb);      // 32768 waves, 32x64 tiles
    k_sim_topk<<<512, 256, 0, stream>>>(qb, keysb, tks, tki);
    k_merge   <<<(BS_ * H_) / 256, 256, 0, stream>>>(tks, tki, outs, outi);
}
